// gnn_15960098471965
// MI455X (gfx1250) — matrix operand order repaired and checked by reading, not compiled, not run on hardware
//
#include <hip/hip_runtime.h>
#include <hip/hip_bf16.h>

// ---------------------------------------------------------------------------
// GraphSAGE link prediction for MI455X (gfx1250), bf16 WMMA GEMMs.
// Ping-pong software-pipelined WMMA loops, global (not flat) loads.
// ---------------------------------------------------------------------------

typedef __attribute__((ext_vector_type(16))) __bf16 v16bf;
typedef __attribute__((ext_vector_type(8)))  float  v8f;
typedef __attribute__((ext_vector_type(2)))  __bf16 bf16x2;
typedef __attribute__((ext_vector_type(8)))  __bf16 bf16x8;

#define NN   20000
#define NE   160000
#define DD   512
#define HID  128

// ----- small utility kernels ------------------------------------------------

__global__ __launch_bounds__(256) void zero_f32_kernel(float* p, long n) {
    long i = (long)blockIdx.x * blockDim.x + threadIdx.x;
    if (i < n) p[i] = 0.0f;
}

__global__ __launch_bounds__(256) void zero_i32_kernel(int* p, int n) {
    int i = blockIdx.x * blockDim.x + threadIdx.x;
    if (i < n) p[i] = 0;
}

__global__ __launch_bounds__(256) void deg_kernel(const int* __restrict__ dst,
                                                  int* __restrict__ deg, int ne) {
    int e = blockIdx.x * blockDim.x + threadIdx.x;
    if (e < ne) atomicAdd(&deg[dst[e]], 1);
}

__global__ __launch_bounds__(256) void invdeg_kernel(const int* __restrict__ deg,
                                                     float* __restrict__ inv, int n) {
    int i = blockIdx.x * blockDim.x + threadIdx.x;
    if (i < n) inv[i] = 1.0f / fmaxf((float)deg[i], 1.0f);
}

// fp32 -> bf16, optional per-row scale (used for msg * inv_deg)
__global__ __launch_bounds__(256) void cvt_bf16_kernel(const float* __restrict__ s,
                                                       const float* __restrict__ rowScale,
                                                       __bf16* __restrict__ d,
                                                       long n, int cols) {
    long i = (long)blockIdx.x * blockDim.x + threadIdx.x;
    if (i >= n) return;
    float v = s[i];
    if (rowScale) v *= rowScale[i / cols];
    d[i] = (__bf16)v;
}

// W[K][N] fp32 -> Wt[N][K] bf16 (so B fragments load contiguous K-pairs)
__global__ __launch_bounds__(256) void transpose_cvt_kernel(const float* __restrict__ W,
                                                            __bf16* __restrict__ Wt,
                                                            int K, int N) {
    int i = blockIdx.x * blockDim.x + threadIdx.x;
    if (i >= K * N) return;
    int k = i / N, n = i % N;
    Wt[(size_t)n * K + k] = (__bf16)W[i];
}

// msg[dst] += x_bf[src]  (fp32 atomics, 8 feats per thread)
__global__ __launch_bounds__(256) void scatter_add_kernel(const __bf16* __restrict__ xb,
                                                          const int* __restrict__ src,
                                                          const int* __restrict__ dst,
                                                          float* __restrict__ msg) {
    long tid = (long)blockIdx.x * blockDim.x + threadIdx.x;   // NE*64 threads
    int e = (int)(tid >> 6);
    int c = ((int)tid & 63) * 8;
    int s = src[e];
    int d = dst[e];
    bf16x8 v = *(const bf16x8*)(xb + (size_t)s * DD + c);
    float* mp = msg + (size_t)d * DD + c;
#pragma unroll
    for (int i = 0; i < 8; ++i)
        atomicAdd(mp + i, (float)v[i]);
}

// ----- WMMA fragment loaders (per ISA 7.12.2 16-bit layouts) ----------------
// A 16x32 (MxK): lane = hh*16 + m ; VGPR j<4: K = 8*hh + 2j ; j>=4: K = 16 + 8*hh + 2(j-4)
__device__ inline v16bf load_a_frag(const __bf16* __restrict__ rowbase, int hh) {
    v16bf a;
#pragma unroll
    for (int j = 0; j < 8; ++j) {
        int kk = (j < 4) ? (8 * hh + 2 * j) : (16 + 8 * hh + 2 * (j - 4));
        bf16x2 p = *(const bf16x2*)(rowbase + kk);
        a[2 * j] = p.x; a[2 * j + 1] = p.y;
    }
    return a;
}
// B 32x16 (KxN), W stored transposed [N][K]: lane = hh*16 + n ; VGPR j<4: K = 8*hh + 2j ; j>=4: K = 16 + 8*hh + 2(j-4)
__device__ inline v16bf load_b_frag(const __bf16* __restrict__ colbase, int hh) {
    v16bf b;
#pragma unroll
    for (int j = 0; j < 8; ++j) {
        int kk = (j < 4) ? (8 * hh + 2 * j) : (16 + 8 * hh + 2 * (j - 4));
        bf16x2 p = *(const bf16x2*)(colbase + kk);
        b[2 * j] = p.x; b[2 * j + 1] = p.y;
    }
    return b;
}

__device__ inline v8f wmma_bf16(v16bf a, v16bf b, v8f c) {
    return __builtin_amdgcn_wmma_f32_16x16x32_bf16(false, a, false, b,
                                                   (short)0, c, false, false);
}

// ----- SAGE conv GEMM: out = relu?(A1@W1 + A2@W2 + bias) --------------------
// Block: 256 threads (8 waves) covers 32 rows x 512 cols.
// Wave w owns cols [w*64, w*64+64) over 2 M-tiles; 8 WMMAs per k-step.
// Two phases (A1/W1t then A2/W2t), 16 k-steps of 32 each, ping-pong
// double-buffered fragments with a cross-phase prefetch so the pipeline
// never drains. All load sites see statically-known global pointers.
__global__ __launch_bounds__(256) void sage_gemm_kernel(
    const __bf16* __restrict__ A1, const __bf16* __restrict__ A2,
    const __bf16* __restrict__ W1t, const __bf16* __restrict__ W2t,
    const float* __restrict__ bias, float* __restrict__ outp, int do_relu) {
    const int lane   = threadIdx.x & 31;
    const int wave   = threadIdx.x >> 5;
    const int lane16 = lane & 15;
    const int hh     = lane >> 4;
    const int m0     = blockIdx.x * 32;
    const int n0     = wave * 64;

    v8f acc[2][4] = {};

    auto load_step = [&](const __bf16* __restrict__ A, const __bf16* __restrict__ Wt,
                         int kb, v16bf& a0, v16bf& a1, v16bf (&bb)[4]) {
        const int k0 = kb * 32;
        a0 = load_a_frag(A + (size_t)(m0 + lane16) * DD + k0, hh);
        a1 = load_a_frag(A + (size_t)(m0 + 16 + lane16) * DD + k0, hh);
#pragma unroll
        for (int t = 0; t < 4; ++t)
            bb[t] = load_b_frag(Wt + (size_t)(n0 + t * 16 + lane16) * DD + k0, hh);
    };
    auto compute = [&](v16bf& a0, v16bf& a1, v16bf (&bb)[4]) {
#pragma unroll
        for (int t = 0; t < 4; ++t) {
            acc[0][t] = wmma_bf16(a0, bb[t], acc[0][t]);
            acc[1][t] = wmma_bf16(a1, bb[t], acc[1][t]);
        }
    };

    v16bf pa0, pa1, pb[4];   // ping
    v16bf qa0, qa1, qb[4];   // pong

    // ---- phase 1: A1 @ W1t ----
    load_step(A1, W1t, 0, pa0, pa1, pb);
    for (int i = 0; i < 7; ++i) {
        load_step(A1, W1t, 2 * i + 1, qa0, qa1, qb);
        compute(pa0, pa1, pb);
        load_step(A1, W1t, 2 * i + 2, pa0, pa1, pb);
        compute(qa0, qa1, qb);
    }
    load_step(A1, W1t, 15, qa0, qa1, qb);
    compute(pa0, pa1, pb);
    load_step(A2, W2t, 0, pa0, pa1, pb);   // cross-phase prefetch
    compute(qa0, qa1, qb);

    // ---- phase 2: A2 @ W2t ----
    for (int i = 0; i < 7; ++i) {
        load_step(A2, W2t, 2 * i + 1, qa0, qa1, qb);
        compute(pa0, pa1, pb);
        load_step(A2, W2t, 2 * i + 2, pa0, pa1, pb);
        compute(qa0, qa1, qb);
    }
    load_step(A2, W2t, 15, qa0, qa1, qb);
    compute(pa0, pa1, pb);
    compute(qa0, qa1, qb);

#pragma unroll
    for (int mt = 0; mt < 2; ++mt) {
#pragma unroll
        for (int t = 0; t < 4; ++t) {
            int n = n0 + t * 16 + lane16;
            float bv = bias[n];
#pragma unroll
            for (int r = 0; r < 8; ++r) {      // C/D: VGPR r -> row base + r + 8*hh
                int m = m0 + mt * 16 + r + 8 * hh;
                float v = acc[mt][t][r] + bv;
                if (do_relu) v = fmaxf(v, 0.0f);
                outp[(size_t)m * DD + n] = v;
            }
        }
    }
}

// ----- Edge MLP: e = concat(h[u],h[v]) -> relu(W1) -> relu(W2) -> W3 --------
// grid = E/16 blocks of 256 threads; wave w owns hidden cols [w*16, w*16+16)
__global__ __launch_bounds__(256) void edge_mlp_kernel(
    const __bf16* __restrict__ hfeat,
    const int* __restrict__ U, const int* __restrict__ V,
    const __bf16* __restrict__ W1t, const float* __restrict__ b1,
    const __bf16* __restrict__ W2t, const float* __restrict__ b2,
    const float* __restrict__ W3, const float* __restrict__ b3,
    float* __restrict__ outp) {
    __shared__ __bf16 act1[16][HID + 8];
    __shared__ __bf16 act2[16][HID + 8];
    __shared__ int uidx[16];
    __shared__ int vidx[16];

    const int e0 = blockIdx.x * 16;
    if (threadIdx.x < 16)      uidx[threadIdx.x]      = U[e0 + threadIdx.x];
    else if (threadIdx.x < 32) vidx[threadIdx.x - 16] = V[e0 + threadIdx.x - 16];
    __syncthreads();

    const int lane   = threadIdx.x & 31;
    const int wave   = threadIdx.x >> 5;
    const int lane16 = lane & 15;
    const int hh     = lane >> 4;
    const int n      = wave * 16 + lane16;       // hidden column 0..127
    const int nu     = uidx[lane16];
    const int nv     = vidx[lane16];

    // layer 1: [16 x 1024] @ [1024 x 128]; A gathered from h via u (k<512) / v.
    // 32 k-steps, ping-pong double-buffered.
    auto load1 = [&](int kb, v16bf& a, v16bf& b) {
        const int k0 = kb * 32;
        const int node = (k0 < DD) ? nu : nv;
        a = load_a_frag(hfeat + (size_t)node * DD + (k0 & (DD - 1)), hh);
        b = load_b_frag(W1t + (size_t)n * (2 * DD) + k0, hh);
    };
    v8f acc = {};
    {
        v16bf pa, pb, qa, qb;
        load1(0, pa, pb);
        for (int i = 0; i < 15; ++i) {
            load1(2 * i + 1, qa, qb);
            acc = wmma_bf16(pa, pb, acc);
            load1(2 * i + 2, pa, pb);
            acc = wmma_bf16(qa, qb, acc);
        }
        load1(31, qa, qb);
        acc = wmma_bf16(pa, pb, acc);
        acc = wmma_bf16(qa, qb, acc);
    }
    {
        float bv = b1[n];
#pragma unroll
        for (int r = 0; r < 8; ++r)
            act1[r + 8 * hh][n] = (__bf16)fmaxf(acc[r] + bv, 0.0f);
    }
    __syncthreads();

    // layer 2: [16 x 128] @ [128 x 128]
    v8f acc2 = {};
#pragma unroll
    for (int kb = 0; kb < HID / 32; ++kb) {
        const int k0 = kb * 32;
        v16bf a = load_a_frag(&act1[lane16][k0], hh);
        v16bf b = load_b_frag(W2t + (size_t)n * HID + k0, hh);
        acc2 = wmma_bf16(a, b, acc2);
    }
    {
        float bv = b2[n];
#pragma unroll
        for (int r = 0; r < 8; ++r)
            act2[r + 8 * hh][n] = (__bf16)fmaxf(acc2[r] + bv, 0.0f);
    }
    __syncthreads();

    // layer 3: [16 x 128] @ [128 x 2] (tiny, scalar fp32)
    if (threadIdx.x < 32) {
        int m = threadIdx.x >> 1;
        int c = threadIdx.x & 1;
        float s = b3[c];
        for (int k = 0; k < HID; ++k)
            s += (float)act2[m][k] * W3[k * 2 + c];
        outp[(size_t)(e0 + m) * 2 + c] = s;
    }
}

// ---------------------------------------------------------------------------

extern "C" void kernel_launch(void* const* d_in, const int* in_sizes, int n_in,
                              void* d_out, int out_size, void* d_ws, size_t ws_size,
                              hipStream_t stream) {
    (void)in_sizes; (void)n_in; (void)out_size; (void)ws_size;

    const float* x       = (const float*)d_in[0];
    const float* W_self  = (const float*)d_in[1];   // [3][512][512]
    const float* W_neigh = (const float*)d_in[2];   // [3][512][512]
    const float* b       = (const float*)d_in[3];   // [3][512]
    const float* W1      = (const float*)d_in[4];   // [1024][128]
    const float* b1      = (const float*)d_in[5];
    const float* W2      = (const float*)d_in[6];   // [128][128]
    const float* b2      = (const float*)d_in[7];
    const float* W3      = (const float*)d_in[8];   // [128][2]
    const float* b3      = (const float*)d_in[9];
    const int* src       = (const int*)d_in[10];
    const int* dst       = (const int*)d_in[11];
    const int* pos_src   = (const int*)d_in[12];
    const int* pos_dst   = (const int*)d_in[13];
    const int* neg_src   = (const int*)d_in[14];
    const int* neg_dst   = (const int*)d_in[15];
    float* out = (float*)d_out;

    // ---- workspace carve (256B aligned) ----
    char* ws = (char*)d_ws;
    size_t off = 0;
    auto carve = [&](size_t bytes) -> void* {
        off = (off + 255) & ~(size_t)255;
        void* p = ws + off;
        off += bytes;
        return p;
    };
    int*    deg    = (int*)   carve((size_t)NN * 4);
    float*  invd   = (float*) carve((size_t)NN * 4);
    float*  msgf   = (float*) carve((size_t)NN * DD * 4);  // msg fp32, later GEMM out
    __bf16* xbf0   = (__bf16*)carve((size_t)NN * DD * 2);
    __bf16* xbf1   = (__bf16*)carve((size_t)NN * DD * 2);
    __bf16* msgbf  = (__bf16*)carve((size_t)NN * DD * 2);
    __bf16* wts    = (__bf16*)carve((size_t)DD * DD * 2);
    __bf16* wtn    = (__bf16*)carve((size_t)DD * DD * 2);
    __bf16* w1t    = (__bf16*)carve((size_t)HID * 2 * DD * 2);
    __bf16* w2t    = (__bf16*)carve((size_t)HID * HID * 2);

    const long nfeat = (long)NN * DD;
    const int  featBlocks = (int)((nfeat + 255) / 256);

    // degrees
    zero_i32_kernel<<<(NN + 255) / 256, 256, 0, stream>>>(deg, NN);
    deg_kernel<<<(NE + 255) / 256, 256, 0, stream>>>(dst, deg, NE);
    invdeg_kernel<<<(NN + 255) / 256, 256, 0, stream>>>(deg, invd, NN);

    // MLP weight transposes to bf16 (once)
    transpose_cvt_kernel<<<(2 * DD * HID + 255) / 256, 256, 0, stream>>>(W1, w1t, 2 * DD, HID);
    transpose_cvt_kernel<<<(HID * HID + 255) / 256, 256, 0, stream>>>(W2, w2t, HID, HID);

    // x -> bf16
    cvt_bf16_kernel<<<featBlocks, 256, 0, stream>>>(x, nullptr, xbf0, nfeat, DD);

    __bf16* cur = xbf0;
    __bf16* nxt = xbf1;
    for (int i = 0; i < 3; ++i) {
        transpose_cvt_kernel<<<(DD * DD + 255) / 256, 256, 0, stream>>>(
            W_self + (size_t)i * DD * DD, wts, DD, DD);
        transpose_cvt_kernel<<<(DD * DD + 255) / 256, 256, 0, stream>>>(
            W_neigh + (size_t)i * DD * DD, wtn, DD, DD);

        zero_f32_kernel<<<featBlocks, 256, 0, stream>>>(msgf, nfeat);
        scatter_add_kernel<<<NE * 64 / 256, 256, 0, stream>>>(cur, src, dst, msgf);
        cvt_bf16_kernel<<<featBlocks, 256, 0, stream>>>(msgf, invd, msgbf, nfeat, DD);

        sage_gemm_kernel<<<NN / 32, 256, 0, stream>>>(
            cur, msgbf, wts, wtn, b + (size_t)i * DD, msgf, (i < 2) ? 1 : 0);

        cvt_bf16_kernel<<<featBlocks, 256, 0, stream>>>(msgf, nullptr, nxt, nfeat, DD);
        __bf16* t = cur; cur = nxt; nxt = t;
    }
    // cur now holds h in bf16

    edge_mlp_kernel<<<NE / 16, 256, 0, stream>>>(
        cur, pos_src, pos_dst, w1t, b1, w2t, b2, W3, b3, out);
    edge_mlp_kernel<<<NE / 16, 256, 0, stream>>>(
        cur, neg_src, neg_dst, w1t, b1, w2t, b2, W3, b3, out + (size_t)2 * NE);
}
